// MultiHeadAttention_34643206210010
// MI455X (gfx1250) — matrix operand order repaired and checked by reading, not compiled, not run on hardware
//
#include <hip/hip_runtime.h>

// ---------------------------------------------------------------------------
// MI455X (gfx1250, wave32) banded multi-head attention + residual + LayerNorm
// All GEMM-shaped math on v_wmma_f32_16x16x32_f16 (f16 in, f32 accumulate).
// Weights pre-transposed to f16 [N][K] so all GEMM tile staging is pure
// b128 traffic (no scalar LDS stores, no per-iter cvt for B).
// ---------------------------------------------------------------------------

typedef __attribute__((ext_vector_type(16))) _Float16 v16h;
typedef __attribute__((ext_vector_type(8)))  _Float16 v8h;
typedef __attribute__((ext_vector_type(8)))  float    v8f;
typedef __attribute__((ext_vector_type(4)))  float    f4;

union AFrag { v16h v; _Float16 e[16]; v8h h8[2]; };
union CFrag { v8f  v; float    e[8];  };
union H8    { v8h  v; _Float16 e[8];  };

#define WMMA_F32_F16(a, b, c) \
    __builtin_amdgcn_wmma_f32_16x16x32_f16(false, (a), false, (b), (short)0, (c), false, false)

namespace cfg {
constexpr int D_MODEL = 512;
constexpr int N_HEADS = 8;
constexpr int DK      = 64;
constexpr int SEQ     = 2048;
constexpr int BATCH   = 4;
constexpr int WINDOW  = 256;
constexpr int BPAD    = 40;          // padded LDS inner dim (80B rows)
}

// ---------------------------------------------------------------------------
// Weight transpose: W[K][N] (f32) -> Wt[N][K] (f16).  Tiled, coalesced.
// ---------------------------------------------------------------------------
__global__ __launch_bounds__(256) void transpose_w_f16(
    const float* __restrict__ W, _Float16* __restrict__ Wt, int K, int N)
{
    __shared__ float tile[32][33];
    const int tx = threadIdx.x & 31;
    const int ty = threadIdx.x >> 5;           // 0..7
    const int k0 = blockIdx.y * 32;
    const int n0 = blockIdx.x * 32;
#pragma unroll
    for (int r = ty; r < 32; r += 8)
        tile[r][tx] = W[(size_t)(k0 + r) * N + n0 + tx];
    __syncthreads();
#pragma unroll
    for (int r = ty; r < 32; r += 8)
        Wt[(size_t)(n0 + r) * K + k0 + tx] = (_Float16)tile[tx][r];
}

// ---------------------------------------------------------------------------
// Projection GEMM: C[M,N] = A[M,K](f32) * Wt[N,K](f16, pre-transposed).
// Block tile 64x128, 8 waves, wave tile 32x32 (4 WMMAs / K-chunk).
// TRANS_OUT=false: row-major C[r*N + c] (Q/K projections).
// TRANS_OUT=true : Vt[(b*512 + c)*SEQ + s] (V projection, key-contiguous).
// ---------------------------------------------------------------------------
template <bool TRANS_OUT>
__global__ __launch_bounds__(256) void gemm_proj(
    const float* __restrict__ A, const _Float16* __restrict__ Wt,
    _Float16* __restrict__ C, int M, int N, int K)
{
    using namespace cfg;
    __shared__ _Float16 As[64][BPAD];      // A tile 64x32 (f16)
    __shared__ _Float16 Bt[128][BPAD];     // B tile: Bt[n][k]

    const int tid  = threadIdx.x;
    const int lane = tid & 31;
    const int wave = tid >> 5;       // 0..7
    const int wm   = wave & 1;       // 32-row subtile
    const int wn   = wave >> 1;      // 32-col subtile (0..3)
    const int m0   = blockIdx.y * 64;
    const int n0   = blockIdx.x * 128;
    const int nlo  = lane & 15;
    const int hi16 = lane >> 4;

    CFrag c[2][2];
#pragma unroll
    for (int s = 0; s < 2; ++s)
#pragma unroll
        for (int t = 0; t < 2; ++t)
#pragma unroll
            for (int v = 0; v < 8; ++v) c[s][t].e[v] = 0.f;

    // staging coords: A -> thread loads 8 contiguous floats of one row
    const int ar = tid >> 2;              // 0..63
    const int ac = (tid & 3) << 3;        // 0,8,16,24
    // B -> thread copies 16 contiguous f16 of one Wt row (two v8h)
    const int br = tid >> 1;              // 0..127
    const int bc = (tid & 1) << 4;        // 0 or 16

    for (int k0 = 0; k0 < K; k0 += 32) {
        if (k0 + 32 < K) {
            __builtin_prefetch(&A[(size_t)(m0 + ar) * K + k0 + 32 + ac], 0, 1);
            __builtin_prefetch(&Wt[(size_t)(n0 + br) * K + k0 + 32 + bc], 0, 1);
        }
        {   // A tile: two float4 loads -> one 16B LDS store (with cvt)
            const float* ap = &A[(size_t)(m0 + ar) * K + k0 + ac];
            f4 x0 = *(const f4*)ap;
            f4 x1 = *(const f4*)(ap + 4);
            H8 p;
#pragma unroll
            for (int j = 0; j < 4; ++j) { p.e[j] = (_Float16)x0[j]; p.e[4 + j] = (_Float16)x1[j]; }
            *(v8h*)&As[ar][ac] = p.v;
        }
        {   // B tile: pure 16B copies from pre-transposed f16 weights
            const _Float16* wp = &Wt[(size_t)(n0 + br) * K + k0 + bc];
            *(v8h*)&Bt[br][bc]     = *(const v8h*)wp;
            *(v8h*)&Bt[br][bc + 8] = *(const v8h*)(wp + 8);
        }
        __syncthreads();

        AFrag a[2], b[2];
        const int kba = hi16 * 16;
        const int kbb = hi16 * 16;
#pragma unroll
        for (int s = 0; s < 2; ++s) {
            const int row = wm * 32 + s * 16 + nlo;
            a[s].h8[0] = *(const v8h*)&As[row][kba];
            a[s].h8[1] = *(const v8h*)&As[row][kba + 8];
        }
#pragma unroll
        for (int t = 0; t < 2; ++t) {
            const int col = wn * 32 + t * 16 + nlo;
            b[t].h8[0] = *(const v8h*)&Bt[col][kbb];
            b[t].h8[1] = *(const v8h*)&Bt[col][kbb + 8];
        }
#pragma unroll
        for (int s = 0; s < 2; ++s)
#pragma unroll
            for (int t = 0; t < 2; ++t)
                c[s][t].v = WMMA_F32_F16(a[s].v, b[t].v, c[s][t].v);
        __syncthreads();
    }

    const int rowbase = hi16 * 8;
    if (!TRANS_OUT) {
#pragma unroll
        for (int s = 0; s < 2; ++s)
#pragma unroll
            for (int v = 0; v < 8; ++v) {
                size_t r = (size_t)(m0 + wm * 32 + s * 16 + rowbase + v);
#pragma unroll
                for (int t = 0; t < 2; ++t)
                    C[r * N + (n0 + wn * 32 + t * 16 + nlo)] = (_Float16)c[s][t].e[v];
            }
    } else {
        // Vt[(b*512 + col) * SEQ + seqpos]: 8 consecutive seq positions/lane
        const int bb = (m0 + wm * 32) / SEQ;
        const int s0base = (m0 + wm * 32) % SEQ;
#pragma unroll
        for (int s = 0; s < 2; ++s)
#pragma unroll
            for (int t = 0; t < 2; ++t) {
                const int col = n0 + wn * 32 + t * 16 + nlo;
                H8 p;
#pragma unroll
                for (int v = 0; v < 8; ++v) p.e[v] = (_Float16)c[s][t].e[v];
                *(v8h*)&C[((size_t)bb * 512 + col) * SEQ + s0base + s * 16 + rowbase] = p.v;
            }
    }
}

// ---------------------------------------------------------------------------
// FC GEMM + residual: C[M,N](f32) = A[M,K](f16) * Wt[N,K](f16) + R[M,N](f32)
// ---------------------------------------------------------------------------
__global__ __launch_bounds__(256) void gemm_fc_residual(
    const _Float16* __restrict__ A, const _Float16* __restrict__ Wt,
    const float* __restrict__ R, float* __restrict__ C, int M, int N, int K)
{
    using namespace cfg;
    __shared__ _Float16 As[64][BPAD];
    __shared__ _Float16 Bt[128][BPAD];

    const int tid  = threadIdx.x;
    const int lane = tid & 31;
    const int wave = tid >> 5;
    const int wm   = wave & 1;
    const int wn   = wave >> 1;
    const int m0   = blockIdx.y * 64;
    const int n0   = blockIdx.x * 128;
    const int nlo  = lane & 15;
    const int hi16 = lane >> 4;

    CFrag c[2][2];
#pragma unroll
    for (int s = 0; s < 2; ++s)
#pragma unroll
        for (int t = 0; t < 2; ++t)
#pragma unroll
            for (int v = 0; v < 8; ++v) c[s][t].e[v] = 0.f;

    const int ar = tid >> 2;
    const int ac = (tid & 3) << 3;
    const int br = tid >> 1;
    const int bc = (tid & 1) << 4;

    for (int k0 = 0; k0 < K; k0 += 32) {
        if (k0 + 32 < K) {
            __builtin_prefetch(&A[(size_t)(m0 + ar) * K + k0 + 32 + ac], 0, 1);
            __builtin_prefetch(&Wt[(size_t)(n0 + br) * K + k0 + 32 + bc], 0, 1);
        }
        // A already f16: one 16B global load -> one 16B LDS store
        *(v8h*)&As[ar][ac] = *(const v8h*)&A[(size_t)(m0 + ar) * K + k0 + ac];
        {
            const _Float16* wp = &Wt[(size_t)(n0 + br) * K + k0 + bc];
            *(v8h*)&Bt[br][bc]     = *(const v8h*)wp;
            *(v8h*)&Bt[br][bc + 8] = *(const v8h*)(wp + 8);
        }
        __syncthreads();

        AFrag a[2], b[2];
        const int kba = hi16 * 16;
        const int kbb = hi16 * 16;
#pragma unroll
        for (int s = 0; s < 2; ++s) {
            const int row = wm * 32 + s * 16 + nlo;
            a[s].h8[0] = *(const v8h*)&As[row][kba];
            a[s].h8[1] = *(const v8h*)&As[row][kba + 8];
        }
#pragma unroll
        for (int t = 0; t < 2; ++t) {
            const int col = wn * 32 + t * 16 + nlo;
            b[t].h8[0] = *(const v8h*)&Bt[col][kbb];
            b[t].h8[1] = *(const v8h*)&Bt[col][kbb + 8];
        }
#pragma unroll
        for (int s = 0; s < 2; ++s)
#pragma unroll
            for (int t = 0; t < 2; ++t)
                c[s][t].v = WMMA_F32_F16(a[s].v, b[t].v, c[s][t].v);
        __syncthreads();
    }

    const int rowbase = hi16 * 8;
#pragma unroll
    for (int s = 0; s < 2; ++s)
#pragma unroll
        for (int v = 0; v < 8; ++v) {
            size_t r = (size_t)(m0 + wm * 32 + s * 16 + rowbase + v);
#pragma unroll
            for (int t = 0; t < 2; ++t) {
                size_t i = r * N + (n0 + wn * 32 + t * 16 + nlo);
                C[i] = c[s][t].e[v] + R[i];
            }
        }
}

// ---------------------------------------------------------------------------
// Banded flash attention. One wave owns a 16-query tile of one (batch, head).
// K fragments: contiguous 32B global loads. V fragments: contiguous 32B loads
// from the pre-transposed Vt. P transpose via per-wave LDS (same-wave DS ops
// are in-order; no block barrier — waves have divergent band lengths).
// ---------------------------------------------------------------------------
__global__ __launch_bounds__(256) void banded_attention(
    const _Float16* __restrict__ Qp, const _Float16* __restrict__ Kp,
    const _Float16* __restrict__ Vt, _Float16* __restrict__ ctx)
{
    using namespace cfg;
    __shared__ _Float16 Plds[8][16][32];   // per-wave P scratch (8 KB)

    const int tid  = threadIdx.x;
    const int lane = tid & 31;
    const int wave = tid >> 5;
    const int b    = blockIdx.z;
    const int h    = blockIdx.y;
    const int q0   = (blockIdx.x * 8 + wave) * 16;
    const int nlo  = lane & 15;
    const int hi16 = lane >> 4;
    const int rowbase = hi16 * 8;

    const size_t basebl = (size_t)b * SEQ * D_MODEL + (size_t)h * DK;

    // Q fragments for d-chunks [0,32) and [32,64): two 16B loads each.
    AFrag aq[2];
    {
        const _Float16* qp = Qp + basebl + (size_t)(q0 + nlo) * D_MODEL + hi16 * 16;
#pragma unroll
        for (int cdx = 0; cdx < 2; ++cdx) {
            aq[cdx].h8[0] = *(const v8h*)(qp + cdx * 32);
            aq[cdx].h8[1] = *(const v8h*)(qp + cdx * 32 + 8);
        }
    }

    float m_run[8], l_run[8];
    CFrag o[4];
#pragma unroll
    for (int v = 0; v < 8; ++v) { m_run[v] = -1e30f; l_run[v] = 0.f; }
#pragma unroll
    for (int j = 0; j < 4; ++j)
#pragma unroll
        for (int v = 0; v < 8; ++v) o[j].e[v] = 0.f;

    int lo = q0 - (WINDOW - 1); if (lo < 0) lo = 0;
    const int kstart = (lo >> 5) << 5;
    const int kend   = q0 + 15;

    // Per-lane invariant pieces of the V^T base address
    const _Float16* vtb = Vt + ((size_t)b * 512 + h * DK + nlo) * SEQ + hi16 * 16;

    for (int ks = kstart; ks <= kend; ks += 32) {
        // ---- scores: two 16x16 tiles over this 32-key chunk -----------------
        CFrag s0, s1;
#pragma unroll
        for (int v = 0; v < 8; ++v) { s0.e[v] = 0.f; s1.e[v] = 0.f; }
        {
            const _Float16* kp0 = Kp + basebl + (size_t)(ks + nlo) * D_MODEL + hi16 * 16;
            const _Float16* kp1 = kp0 + (size_t)16 * D_MODEL;
#pragma unroll
            for (int cdx = 0; cdx < 2; ++cdx) {
                AFrag bk0, bk1;
                bk0.v = *(const v16h*)(kp0 + cdx * 32);
                bk1.v = *(const v16h*)(kp1 + cdx * 32);
                s0.v = WMMA_F32_F16(aq[cdx].v, bk0.v, s0.v);
                s1.v = WMMA_F32_F16(aq[cdx].v, bk1.v, s1.v);
            }
        }
        // ---- mask + scale + row-max (reduce across 16-lane half groups) -----
        float pm[8];
#pragma unroll
        for (int v = 0; v < 8; ++v) {
            const int i  = q0 + rowbase + v;
            const int j0 = ks + nlo, j1 = ks + 16 + nlo;
            float x0 = s0.e[v] * 0.125f;   // 1/sqrt(64)
            float x1 = s1.e[v] * 0.125f;
            if (!(j0 <= i && j0 + (WINDOW - 1) >= i)) x0 = -1e30f;
            if (!(j1 <= i && j1 + (WINDOW - 1) >= i)) x1 = -1e30f;
            s0.e[v] = x0; s1.e[v] = x1;
            float t = fmaxf(x0, x1);
            t = fmaxf(t, __shfl_xor(t, 1));
            t = fmaxf(t, __shfl_xor(t, 2));
            t = fmaxf(t, __shfl_xor(t, 4));
            t = fmaxf(t, __shfl_xor(t, 8));
            pm[v] = t;
        }
        // ---- online softmax update ------------------------------------------
#pragma unroll
        for (int v = 0; v < 8; ++v) {
            const float mnew  = fmaxf(m_run[v], pm[v]);
            const float alpha = __expf(m_run[v] - mnew);
            float p0 = (s0.e[v] <= -1e29f) ? 0.f : __expf(s0.e[v] - mnew);
            float p1 = (s1.e[v] <= -1e29f) ? 0.f : __expf(s1.e[v] - mnew);
            s0.e[v] = p0; s1.e[v] = p1;
            float rs = p0 + p1;
            rs += __shfl_xor(rs, 1);
            rs += __shfl_xor(rs, 2);
            rs += __shfl_xor(rs, 4);
            rs += __shfl_xor(rs, 8);
            l_run[v] = l_run[v] * alpha + rs;
            m_run[v] = mnew;
#pragma unroll
            for (int j = 0; j < 4; ++j) o[j].e[v] *= alpha;
        }
        // ---- P: C-layout -> row-major LDS -> A-fragment (two 16B reads) -----
#pragma unroll
        for (int v = 0; v < 8; ++v) {
            Plds[wave][rowbase + v][nlo]      = (_Float16)s0.e[v];
            Plds[wave][rowbase + v][16 + nlo] = (_Float16)s1.e[v];
        }
        AFrag ap;
        ap.h8[0] = *(const v8h*)&Plds[wave][nlo][hi16 * 16];
        ap.h8[1] = *(const v8h*)&Plds[wave][nlo][hi16 * 16 + 8];
        // ---- O += P * V : V^T rows are key-contiguous -> 32B loads ----------
        {
            const _Float16* vp = vtb + ks;
#pragma unroll
            for (int j = 0; j < 4; ++j) {
                AFrag bv;
                bv.v = *(const v16h*)(vp + (size_t)j * 16 * SEQ);
                o[j].v = WMMA_F32_F16(ap.v, bv.v, o[j].v);
            }
        }
    }

    // ---- normalize & store context (f16) ------------------------------------
#pragma unroll
    for (int v = 0; v < 8; ++v) {
        const float inv = 1.0f / l_run[v];
        const size_t r = (size_t)b * SEQ + q0 + rowbase + v;
#pragma unroll
        for (int j = 0; j < 4; ++j)
            ctx[r * D_MODEL + h * DK + j * 16 + nlo] = (_Float16)(o[j].e[v] * inv);
    }
}

// ---------------------------------------------------------------------------
// In-place LayerNorm over rows of 512 (gamma=1, beta=0, population variance).
// ---------------------------------------------------------------------------
__global__ __launch_bounds__(256) void layernorm_inplace(float* __restrict__ X, int N)
{
    __shared__ float reds[8], redq[8], mv[2];
    float* x = X + (size_t)blockIdx.x * N;
    const int tid = threadIdx.x;

    float v0 = x[tid], v1 = x[tid + 256];
    float s  = v0 + v1;
    float ss = v0 * v0 + v1 * v1;
#pragma unroll
    for (int m = 1; m < 32; m <<= 1) { s += __shfl_xor(s, m); ss += __shfl_xor(ss, m); }
    if ((tid & 31) == 0) { reds[tid >> 5] = s; redq[tid >> 5] = ss; }
    __syncthreads();
    if (tid == 0) {
        float ts = 0.f, tq = 0.f;
#pragma unroll
        for (int i = 0; i < 8; ++i) { ts += reds[i]; tq += redq[i]; }
        const float mean = ts / N;
        const float var  = tq / N - mean * mean;
        mv[0] = mean;
        mv[1] = rsqrtf(var + 1e-5f);
    }
    __syncthreads();
    const float mean = mv[0], rstd = mv[1];
    x[tid]       = (v0 - mean) * rstd;
    x[tid + 256] = (v1 - mean) * rstd;
}

// ---------------------------------------------------------------------------
extern "C" void kernel_launch(void* const* d_in, const int* in_sizes, int n_in,
                              void* d_out, int out_size, void* d_ws, size_t ws_size,
                              hipStream_t stream)
{
    using namespace cfg;
    const float* inQ = (const float*)d_in[0];
    const float* inK = (const float*)d_in[1];
    const float* inV = (const float*)d_in[2];
    const float* WQ  = (const float*)d_in[3];
    const float* WK  = (const float*)d_in[4];
    const float* WV  = (const float*)d_in[5];
    const float* Wfc = (const float*)d_in[6];

    char* ws = (char*)d_ws;
    const size_t MROWS    = (size_t)BATCH * SEQ;                        // 8192
    const size_t half_mat = MROWS * D_MODEL * sizeof(_Float16);         // 8 MB
    const size_t w_mat    = (size_t)D_MODEL * D_MODEL * sizeof(_Float16); // 512 KB
    _Float16* Qp   = (_Float16*)(ws);
    _Float16* Kp   = (_Float16*)(ws + half_mat);
    _Float16* Vtp  = (_Float16*)(ws + 2 * half_mat);  // transposed: [B*512][SEQ]
    _Float16* ctx  = (_Float16*)(ws + 3 * half_mat);
    _Float16* WQt  = (_Float16*)(ws + 4 * half_mat);
    _Float16* WKt  = (_Float16*)(ws + 4 * half_mat + w_mat);
    _Float16* WVt  = (_Float16*)(ws + 4 * half_mat + 2 * w_mat);
    _Float16* Wfct = (_Float16*)(ws + 4 * half_mat + 3 * w_mat);
    float*  fcout = (float*)d_out;   // pre-LN result lives in d_out, LN in place

    dim3 gtr(D_MODEL / 32, D_MODEL / 32);                               // (16,16)
    transpose_w_f16<<<gtr, 256, 0, stream>>>(WQ,  WQt,  D_MODEL, D_MODEL);
    transpose_w_f16<<<gtr, 256, 0, stream>>>(WK,  WKt,  D_MODEL, D_MODEL);
    transpose_w_f16<<<gtr, 256, 0, stream>>>(WV,  WVt,  D_MODEL, D_MODEL);
    transpose_w_f16<<<gtr, 256, 0, stream>>>(Wfc, Wfct, D_MODEL, D_MODEL);

    dim3 gblk(D_MODEL / 128, (unsigned)(MROWS / 64));                   // (4,128)
    gemm_proj<false><<<gblk, 256, 0, stream>>>(inQ, WQt, Qp,  (int)MROWS, D_MODEL, D_MODEL);
    gemm_proj<false><<<gblk, 256, 0, stream>>>(inK, WKt, Kp,  (int)MROWS, D_MODEL, D_MODEL);
    gemm_proj<true ><<<gblk, 256, 0, stream>>>(inV, WVt, Vtp, (int)MROWS, D_MODEL, D_MODEL);

    dim3 gatt(SEQ / 16 / 8, N_HEADS, BATCH);                            // (16,8,4)
    banded_attention<<<gatt, 256, 0, stream>>>(Qp, Kp, Vtp, ctx);

    gemm_fc_residual<<<gblk, 256, 0, stream>>>(ctx, Wfct, inQ, fcout,
                                               (int)MROWS, D_MODEL, D_MODEL);

    layernorm_inplace<<<(unsigned)MROWS, 256, 0, stream>>>(fcout, D_MODEL);
}
